// ALIGNNAtomWise_90022514524470
// MI455X (gfx1250) — matrix operand order repaired and checked by reading, not compiled, not run on hardware
//
#include <hip/hip_runtime.h>
#include <hip/hip_bf16.h>

// ---------------------------------------------------------------------------
// ALIGNN forward for MI455X (gfx1250, wave32, WMMA bf16, LDS-staged operands)
// ---------------------------------------------------------------------------

typedef __attribute__((ext_vector_type(16))) __bf16 v16bf;
typedef __attribute__((ext_vector_type(8)))  float  v8f;

constexpr int HD = 64;   // hidden dim everywhere

__device__ __forceinline__ float wave_sum32(float v) {
#pragma unroll
  for (int off = 16; off > 0; off >>= 1) v += __shfl_xor(v, off, 32);
  return v;
}
__device__ __forceinline__ float silu_f(float u) { return u / (1.f + __expf(-u)); }
__device__ __forceinline__ float sigm_f(float u) { return 1.f / (1.f + __expf(-u)); }
__device__ __forceinline__ float2 ld2(const float* p) { return *(const float2*)p; }
__device__ __forceinline__ void   st2(float* p, float2 v) { *(float2*)p = v; }

// A-operand source modes for the fused GEMM
#define AM_X     0   // A = X[M x K] row-major f32
#define AM_RBF_E 1   // A = rbf(norm(r_row), 0, 8, 80)
#define AM_RBF_A 2   // A = rbf(angle_cos_row, -1, 1, 40)

// 16-bit A 16x32 WMMA fragment from an LDS row (ISA 7.12.2):
// two contiguous 8-elem bf16 runs per lane -> two ds_load_b128.
template<int AST>
__device__ __forceinline__ v16bf load_afrag(const __bf16* As, int mrow, int khalf, int kt) {
  v16bf af;
#pragma unroll
  for (int j = 0; j < 8; ++j) {
    const int kb = ((j < 4) ? (2 * j) : (16 + 2 * (j - 4))) + 8 * khalf;
    af[2 * j]     = As[mrow * AST + kt + kb];
    af[2 * j + 1] = As[mrow * AST + kt + kb + 1];
  }
  return af;
}
// B 32x16 fragment from transposed-W LDS row: two contiguous 8-elem bf16 runs (k = 8*khalf+s, 16+8*khalf+s-8) -> 2x ds_load_b128.
template<int AST>
__device__ __forceinline__ v16bf load_bfrag(const __bf16* Wt, int nrow, int khalf, int kt) {
  v16bf bfm;
#pragma unroll
  for (int e = 0; e < 16; ++e) bfm[e] = Wt[nrow * AST + kt + 8 * khalf + e + ((e >= 8) ? 8 : 0)];
  return bfm;
}

// ---------------------------------------------------------------------------
// Fused GEMM: Out[M x 64] = act( A[M x K] @ W[K x 64] + bias )
// act = identity (LNSILU=false) or SiLU(LayerNorm(.)) (LNSILU=true)
// Block = 128 threads (4 waves); 16-row tile x 64 cols; K compile-time.
// Epilogue goes through LDS -> coalesced float2 stores; full-tile fast path.
// ---------------------------------------------------------------------------
template<int AMODE, int K, bool LNSILU>
__global__ __launch_bounds__(128)
void gemm64_kernel(const float* __restrict__ X, int M,
                   const float* __restrict__ W,     // K x 64 row-major
                   const float* __restrict__ bias,  // 64
                   const float* __restrict__ lng,   // LN gamma (LNSILU only)
                   const float* __restrict__ lnb,   // LN beta
                   float* __restrict__ Out)         // M x 64
{
  constexpr int Kpad = (K + 31) & ~31;
  constexpr int AST  = 104;                // padded LDS stride (bf16), 16B-aligned
  __shared__ __bf16 As[16 * AST];
  __shared__ __bf16 Wt[64 * AST];
  __shared__ float  Cs[16][HD + 2];        // +2 pad: float2-aligned, conflict-free

  const int tid  = threadIdx.x;
  const int wave = tid >> 5;
  const int lane = tid & 31;
  const int row0 = blockIdx.x * 16;

  if constexpr (AMODE == AM_X) {           // speculative prefetch of next row tile
    if (row0 + 16 < M) __builtin_prefetch(X + (size_t)(row0 + 16) * K + (size_t)tid * 16, 0, 1);
  }

  // --- Stage A tile (16 x Kpad) as bf16, branch-free (clamp + select) ------
  for (int i = tid; i < 16 * Kpad; i += 128) {
    const int rr = i / Kpad;
    const int kk = i - rr * Kpad;
    const int grow = row0 + rr;
    const int ga = (grow < M) ? grow : (M - 1);
    float v;
    if constexpr (AMODE == AM_X) {
      const int ka = (kk < K) ? kk : (K - 1);
      v = X[(size_t)ga * K + ka];
    } else if constexpr (AMODE == AM_RBF_E) {
      const float rx = X[(size_t)ga * 3 + 0];
      const float ry = X[(size_t)ga * 3 + 1];
      const float rz = X[(size_t)ga * 3 + 2];
      const float d  = sqrtf(rx * rx + ry * ry + rz * rz);
      const float t  = d - 8.f * (float)kk / 79.f;         // centers 0..8, 80 bins
      v = __expf(-(79.f / 8.f) * t * t);                    // gamma = 79/8
    } else {
      const float d = X[ga];
      const float t = d - (-1.f + 2.f * (float)kk / 39.f);  // centers -1..1, 40 bins
      v = __expf(-(39.f / 2.f) * t * t);                    // gamma = 39/2
    }
    if (grow >= M || kk >= K) v = 0.f;
    As[rr * AST + kk] = (__bf16)v;
  }
  // --- Stage W transposed (Wt[n][k] = W[k][n]), coalesced global reads -----
  for (int i = tid; i < 64 * Kpad; i += 128) {
    const int kk = i >> 6, n = i & 63;
    const int ka = (kk < K) ? kk : (K - 1);
    float v = W[(size_t)ka * HD + n];
    if (kk >= K) v = 0.f;
    Wt[n * AST + kk] = (__bf16)v;
  }
  __syncthreads();

  const int n0    = wave * 16;
  const int mrow  = lane & 15;
  const int khalf = lane >> 4;

  v8f acc = {};
#pragma unroll
  for (int kt = 0; kt < Kpad; kt += 32) {
    const v16bf af  = load_afrag<AST>(As, mrow, khalf, kt);
    const v16bf bfm = load_bfrag<AST>(Wt, n0 + mrow, khalf, kt);
    acc = __builtin_amdgcn_wmma_f32_16x16x32_bf16(false, af, false, bfm,
                                                  (short)0, acc, false, false);
  }

  // --- Epilogue via LDS: C/D layout VGPR r -> M = r + 8*khalf, N = n0+mrow -
  const float bval = bias[n0 + mrow];
#pragma unroll
  for (int r2 = 0; r2 < 8; ++r2) Cs[r2 + 8 * khalf][n0 + mrow] = acc[r2] + bval;
  __syncthreads();

  // wave handles rows wave*4..+3; lane owns columns 2l, 2l+1 -> float2 stores
  auto row_val = [&](int rr) -> float2 {
    float v0 = Cs[rr][2 * lane];
    float v1 = Cs[rr][2 * lane + 1];
    if constexpr (LNSILU) {
      const float sm = wave_sum32(v0 + v1);
      const float sq = wave_sum32(v0 * v0 + v1 * v1);
      const float mean = sm * (1.f / 64.f);
      const float var  = sq * (1.f / 64.f) - mean * mean;
      const float inv  = rsqrtf(var + 1e-5f);
      const float2 gg = ld2(lng + 2 * lane);
      const float2 bb = ld2(lnb + 2 * lane);
      v0 = silu_f((v0 - mean) * inv * gg.x + bb.x);
      v1 = silu_f((v1 - mean) * inv * gg.y + bb.y);
    }
    return make_float2(v0, v1);
  };

  if (row0 + 16 <= M) {                    // hot path: full tile, no guards
#pragma unroll
    for (int q = 0; q < 4; ++q) {
      const int rr = wave * 4 + q;
      st2(Out + (size_t)(row0 + rr) * HD + 2 * lane, row_val(rr));
    }
  } else {                                 // cold tail
    for (int q = 0; q < 4; ++q) {
      const int rr = wave * 4 + q;
      const float2 v = row_val(rr);
      if (row0 + rr < M) st2(Out + (size_t)(row0 + rr) * HD + 2 * lane, v);
    }
  }
}

// ---------------------------------------------------------------------------
// Fused 4-way projection: Psg/Pdg/Pdu/Psu = X @ {Wsg,Wdg,Wdu,Wsu} + bias.
// 512 threads = 16 waves; wave w -> weight set w/4, column group (w%4)*16.
// ---------------------------------------------------------------------------
__global__ __launch_bounds__(512)
void egc_proj4_kernel(const float* __restrict__ X, int M,
                      const float* __restrict__ Wsg, const float* __restrict__ Bsg,
                      const float* __restrict__ Wdg, const float* __restrict__ Bdg,
                      const float* __restrict__ Wdu, const float* __restrict__ Bdu,
                      const float* __restrict__ Wsu, const float* __restrict__ Bsu,
                      float* __restrict__ Osg, float* __restrict__ Odg,
                      float* __restrict__ Odu, float* __restrict__ Osu)
{
  constexpr int AST = 72;                  // 64 + 8 pad (bf16), 16B-aligned stride
  __shared__ __bf16 As[16 * AST];
  __shared__ __bf16 Wt[4 * 64 * AST];

  const int tid  = threadIdx.x;
  const int wave = tid >> 5;
  const int lane = tid & 31;
  const int set  = wave >> 2;
  const int n0   = (wave & 3) * 16;
  const int row0 = blockIdx.x * 16;

  if (row0 + 16 < M) __builtin_prefetch(X + (size_t)(row0 + 16) * HD + (size_t)tid * 2, 0, 1);

  // Stage A (16 x 64): 2 elements/thread, branch-free
  for (int i = tid; i < 16 * 64; i += 512) {
    const int rr = i >> 6, kk = i & 63;
    const int grow = row0 + rr;
    const int ga = (grow < M) ? grow : (M - 1);
    float v = X[(size_t)ga * HD + kk];
    if (grow >= M) v = 0.f;
    As[rr * AST + kk] = (__bf16)v;
  }
  // Stage 4 weight matrices transposed
#pragma unroll
  for (int s = 0; s < 4; ++s) {
    const float* Wp = (s == 0) ? Wsg : (s == 1) ? Wdg : (s == 2) ? Wdu : Wsu;
    for (int i = tid; i < 64 * 64; i += 512) {
      const int kk = i >> 6, n = i & 63;             // i = kk*64 + n -> Wp[i]
      Wt[(s * 64 + n) * AST + kk] = (__bf16)Wp[i];
    }
  }
  __syncthreads();

  const int mrow  = lane & 15;
  const int khalf = lane >> 4;

  v8f acc = {};
#pragma unroll
  for (int kt = 0; kt < 64; kt += 32) {
    const v16bf af  = load_afrag<AST>(As, mrow, khalf, kt);
    const v16bf bfm = load_bfrag<AST>(Wt, set * 64 + n0 + mrow, khalf, kt);
    acc = __builtin_amdgcn_wmma_f32_16x16x32_bf16(false, af, false, bfm,
                                                  (short)0, acc, false, false);
  }

  const float* Bp = (set == 0) ? Bsg : (set == 1) ? Bdg : (set == 2) ? Bdu : Bsu;
  float*       Op = (set == 0) ? Osg : (set == 1) ? Odg : (set == 2) ? Odu : Osu;
  const float bval = Bp[n0 + mrow];
  float* base = Op + (size_t)row0 * HD + n0 + mrow;
  if (row0 + 16 <= M) {                    // hot path: unguarded stores
#pragma unroll
    for (int r2 = 0; r2 < 8; ++r2)
      base[(size_t)(r2 + 8 * khalf) * HD] = acc[r2] + bval;
  } else {
    for (int r2 = 0; r2 < 8; ++r2) {
      const int grow = row0 + r2 + 8 * khalf;
      if (grow < M) base[(size_t)(r2 + 8 * khalf) * HD] = acc[r2] + bval;
    }
  }
}

// ---------------------------------------------------------------------------
// EGC message-passing: one edge per wave, lane owns features 2l, 2l+1.
// ---------------------------------------------------------------------------
__global__ __launch_bounds__(256)
void egc_edge_kernel(const float* __restrict__ EG,
                     const float* __restrict__ Psg, const float* __restrict__ Pdg,
                     const float* __restrict__ Pdu,
                     float* __restrict__ Y,
                     const int* __restrict__ S, const int* __restrict__ D,
                     float* __restrict__ sumH, float* __restrict__ sumSig,
                     const float* __restrict__ lng, const float* __restrict__ lnb,
                     int Ne)
{
  const int wave = threadIdx.x >> 5;
  const int lane = threadIdx.x & 31;
  const int e = blockIdx.x * 8 + wave;
  if (e >= Ne) return;
  const int s = S[e], d = D[e];
  const size_t eo = (size_t)e * HD + 2 * lane;
  const size_t so = (size_t)s * HD + 2 * lane;
  const size_t dd = (size_t)d * HD + 2 * lane;

  const float2 egv = ld2(EG + eo);
  const float2 sg  = ld2(Psg + so);
  const float2 dg  = ld2(Pdg + dd);
  const float m0 = egv.x + sg.x + dg.x;
  const float m1 = egv.y + sg.y + dg.y;
  const float g0 = sigm_f(m0), g1 = sigm_f(m1);

  const float2 du = ld2(Pdu + so);
  atomicAdd(&sumH[dd],       du.x * g0);
  atomicAdd(&sumH[dd + 1],   du.y * g1);
  atomicAdd(&sumSig[dd],     g0);
  atomicAdd(&sumSig[dd + 1], g1);

  const float sm = wave_sum32(m0 + m1);
  const float sq = wave_sum32(m0 * m0 + m1 * m1);
  const float mean = sm * (1.f / 64.f);
  const float var  = sq * (1.f / 64.f) - mean * mean;
  const float inv  = rsqrtf(var + 1e-5f);
  const float2 gg = ld2(lng + 2 * lane);
  const float2 bb = ld2(lnb + 2 * lane);
  float2 yv = ld2(Y + eo);
  yv.x += silu_f((m0 - mean) * inv * gg.x + bb.x);
  yv.y += silu_f((m1 - mean) * inv * gg.y + bb.y);
  st2(Y + eo, yv);
}

// EGC node finalize: h = sumH/(sumSig+1e-6); X += SiLU(LN(Psu + h))
__global__ __launch_bounds__(256)
void egc_node_kernel(const float* __restrict__ Psu,
                     const float* __restrict__ sumH, const float* __restrict__ sumSig,
                     float* __restrict__ Xio,
                     const float* __restrict__ lng, const float* __restrict__ lnb,
                     int Nn)
{
  const int wave = threadIdx.x >> 5;
  const int lane = threadIdx.x & 31;
  const int n = blockIdx.x * 8 + wave;
  if (n >= Nn) return;
  const size_t no = (size_t)n * HD + 2 * lane;

  const float2 hN = ld2(sumH + no);
  const float2 hS = ld2(sumSig + no);
  const float2 su = ld2(Psu + no);
  const float t0 = su.x + hN.x / (hS.x + 1e-6f);
  const float t1 = su.y + hN.y / (hS.y + 1e-6f);

  const float sm = wave_sum32(t0 + t1);
  const float sq = wave_sum32(t0 * t0 + t1 * t1);
  const float mean = sm * (1.f / 64.f);
  const float var  = sq * (1.f / 64.f) - mean * mean;
  const float inv  = rsqrtf(var + 1e-5f);
  const float2 gg = ld2(lng + 2 * lane);
  const float2 bb = ld2(lnb + 2 * lane);
  float2 xv = ld2(Xio + no);
  xv.x += silu_f((t0 - mean) * inv * gg.x + bb.x);
  xv.y += silu_f((t1 - mean) * inv * gg.y + bb.y);
  st2(Xio + no, xv);
}

__global__ void zero4_kernel(float4* __restrict__ p, size_t n4) {
  size_t i = (size_t)blockIdx.x * blockDim.x + threadIdx.x;
  const size_t stride = (size_t)gridDim.x * blockDim.x;
  for (; i < n4; i += stride) p[i] = make_float4(0.f, 0.f, 0.f, 0.f);
}

__global__ __launch_bounds__(256)
void pool_partial(const float* __restrict__ X, int Nn, float* __restrict__ pooled)
{
  __shared__ float sh[256];
  const int t = threadIdx.x;
  const int f = t & 63, part = t >> 6;
  const int base = blockIdx.x * 256;
  float acc = 0.f;
  for (int nn = part; nn < 256; nn += 4) {
    const int node = base + nn;
    if (node < Nn) acc += X[(size_t)node * HD + f];
  }
  sh[t] = acc;
  __syncthreads();
  if (t < 64) {
    const float tot = sh[t] + sh[t + 64] + sh[t + 128] + sh[t + 192];
    atomicAdd(&pooled[f], tot);
  }
}

__global__ void pool_final(const float* __restrict__ pooled,
                           const float* __restrict__ w, const float* __restrict__ b,
                           float* __restrict__ out, float invN)
{
  const int l = threadIdx.x;
  const float2 pv = ld2(pooled + 2 * l);
  const float2 wv = ld2(w + 2 * l);
  float v = pv.x * wv.x + pv.y * wv.y;
  v = wave_sum32(v);
  if (l == 0) out[0] = v * invN + b[0];
}

// ---------------------------------------------------------------------------
// Host-side dispatch
// ---------------------------------------------------------------------------
template<int AMODE, int K, bool LNSILU>
static inline void launch_gemm(const float* X, int M, const float* W, const float* bias,
                               const float* lng, const float* lnb, float* Out,
                               hipStream_t st)
{
  dim3 grid((unsigned)((M + 15) / 16)), blk(128);
  gemm64_kernel<AMODE, K, LNSILU><<<grid, blk, 0, st>>>(X, M, W, bias, lng, lnb, Out);
}

extern "C" void kernel_launch(void* const* d_in, const int* in_sizes, int n_in,
                              void* d_out, int out_size, void* d_ws, size_t ws_size,
                              hipStream_t stream)
{
  (void)in_sizes; (void)n_in; (void)out_size; (void)ws_size;

  constexpr int N_NODES = 10000, N_EDGES = 80000, N_TRIP = 400000;

  const float* atom  = (const float*)d_in[0];   // 10000 x 92
  const float* rvec  = (const float*)d_in[1];   // 80000 x 3
  const float* acosv = (const float*)d_in[2];   // 400000
  const int*   srcI  = (const int*)d_in[3];
  const int*   dstI  = (const int*)d_in[4];
  const int*   lsrcI = (const int*)d_in[5];
  const int*   ldstI = (const int*)d_in[6];
  float* outp = (float*)d_out;

  auto P = [&](int i) { return (const float*)d_in[i]; };

  // Param leaf layout (JAX sorted-key DFS flatten of params, base index 7):
  //   alignn[L].edge @ 7+L*28, alignn[L].node @ 7+L*28+14  (L=0,1)
  //   each EGC (14 leaves): bnE.b,bnE.g,bnN.b,bnN.g, dg.b,dg.w, du.b,du.w,
  //                         eg.b,eg.w, sg.b,sg.w, su.b,su.w
  //   angle_emb1 @63, angle_emb2 @67, atom_emb @71, edge_emb1 @75,
  //   edge_emb2 @79  (each MLP: lin.b, lin.w, ln.b, ln.g)
  //   fc.b @83, fc.w @84, gcn[0] @85, gcn[1] @99

  // ---- workspace layout ---------------------------------------------------
  char* wsb = (char*)d_ws;
  size_t off = 0;
  auto alloc = [&](size_t bytes) -> float* {
    float* p = (float*)(wsb + off);
    off += (bytes + 255) & ~(size_t)255;
    return p;
  };
  float* Xb   = alloc((size_t)N_NODES * HD * 4);   // node features
  float* Yb   = alloc((size_t)N_EDGES * HD * 4);   // edge features
  float* Zb   = alloc((size_t)N_TRIP  * HD * 4);   // triplet features
  float* Psg  = alloc((size_t)N_EDGES * HD * 4);   // projections (max 80K rows)
  float* Pdg  = alloc((size_t)N_EDGES * HD * 4);
  float* Pdu  = alloc((size_t)N_EDGES * HD * 4);
  float* Psu  = alloc((size_t)N_EDGES * HD * 4);
  float* Mb   = alloc((size_t)N_TRIP  * HD * 4);   // edge-gate / embedding tmp
  float* sumH = alloc((size_t)N_EDGES * HD * 4);
  float* sumS = alloc((size_t)N_EDGES * HD * 4);
  float* pooled = alloc(64 * 4);

  // ---- embeddings (RBF fused into GEMM A-staging) -------------------------
  launch_gemm<AM_X, 92, true>(atom, N_NODES, P(72), P(71), P(74), P(73), Xb, stream);
  launch_gemm<AM_RBF_E, 80, true>(rvec, N_EDGES, P(76), P(75), P(78), P(77), Mb, stream);
  launch_gemm<AM_X, 64, true>(Mb, N_EDGES, P(80), P(79), P(82), P(81), Yb, stream);
  launch_gemm<AM_RBF_A, 40, true>(acosv, N_TRIP, P(64), P(63), P(66), P(65), Mb, stream);
  launch_gemm<AM_X, 64, true>(Mb, N_TRIP, P(68), P(67), P(70), P(69), Zb, stream);

  // ---- one edge-gated graph conv ------------------------------------------
  auto run_egc = [&](int wb, float* Xf, int Nn, float* Yf, int Ne,
                     const int* S, const int* D) {
    const float *bnE_b = P(wb + 0),  *bnE_g = P(wb + 1);
    const float *bnN_b = P(wb + 2),  *bnN_g = P(wb + 3);
    const float *dg_b  = P(wb + 4),  *dg_w  = P(wb + 5);
    const float *du_b  = P(wb + 6),  *du_w  = P(wb + 7);
    const float *eg_b  = P(wb + 8),  *eg_w  = P(wb + 9);
    const float *sg_b  = P(wb + 10), *sg_w  = P(wb + 11);
    const float *su_b  = P(wb + 12), *su_w  = P(wb + 13);

    egc_proj4_kernel<<<dim3((unsigned)((Nn + 15) / 16)), dim3(512), 0, stream>>>(
        Xf, Nn, sg_w, sg_b, dg_w, dg_b, du_w, du_b, su_w, su_b,
        Psg, Pdg, Pdu, Psu);
    launch_gemm<AM_X, 64, false>(Yf, Ne, eg_w, eg_b, nullptr, nullptr, Mb, stream);

    const size_t acc4 = (size_t)Nn * HD / 4;
    zero4_kernel<<<dim3((unsigned)((acc4 + 255) / 256)), dim3(256), 0, stream>>>((float4*)sumH, acc4);
    zero4_kernel<<<dim3((unsigned)((acc4 + 255) / 256)), dim3(256), 0, stream>>>((float4*)sumS, acc4);

    egc_edge_kernel<<<dim3((unsigned)((Ne + 7) / 8)), dim3(256), 0, stream>>>(
        Mb, Psg, Pdg, Pdu, Yf, S, D, sumH, sumS, bnE_g, bnE_b, Ne);
    egc_node_kernel<<<dim3((unsigned)((Nn + 7) / 8)), dim3(256), 0, stream>>>(
        Psu, sumH, sumS, Xf, bnN_g, bnN_b, Nn);
  };

  // ---- ALIGNN layers (node EGC on crystal graph, edge EGC on line graph) --
  for (int L = 0; L < 2; ++L) {
    const int edge_base = 7 + L * 28;
    const int node_base = edge_base + 14;
    run_egc(node_base, Xb, N_NODES, Yb, N_EDGES, srcI, dstI);
    run_egc(edge_base, Yb, N_EDGES, Zb, N_TRIP, lsrcI, ldstI);
  }
  // ---- GCN layers ---------------------------------------------------------
  for (int L = 0; L < 2; ++L) {
    run_egc(85 + L * 14, Xb, N_NODES, Yb, N_EDGES, srcI, dstI);
  }

  // ---- mean pool + fc -----------------------------------------------------
  zero4_kernel<<<1, 64, 0, stream>>>((float4*)pooled, 16);
  pool_partial<<<dim3((unsigned)((N_NODES + 255) / 256)), dim3(256), 0, stream>>>(
      Xb, N_NODES, pooled);
  pool_final<<<1, 32, 0, stream>>>(pooled, P(84), P(83), outp, 1.f / (float)N_NODES);
}
